// GRU_69904887710213
// MI455X (gfx1250) — matrix operand order repaired and checked by reading, not compiled, not run on hardware
//
#include <hip/hip_runtime.h>
#include <hip/hip_bf16.h>

// ---------------- problem dimensions (from reference setup_inputs) ----------
constexpr int T_   = 512;
constexpr int B_   = 64;
constexpr int I_   = 1024;
constexpr int H_   = 1024;
constexpr int G3H  = 3 * H_;        // 3072
constexpr int KDIM = 1024;          // inner dim for both layers (I_ == H_)
constexpr int BH_  = B_ * H_;       // 65536

typedef __attribute__((ext_vector_type(16))) _Float16 v16h;
typedef __attribute__((ext_vector_type(8)))  _Float16 v8h;
typedef __attribute__((ext_vector_type(8)))  float    v8f;
typedef __attribute__((ext_vector_type(4)))  unsigned v4u;
typedef __attribute__((ext_vector_type(8)))  int      v8i;
typedef __attribute__((ext_vector_type(4)))  int      v4i;

// padded LDS row stride for a 32-f16 (64B) tile row, +16B pad -> 80B = 40 f16.
// ds_load_b128 fragment reads at row stride 80B hit banks 20*r mod 64,
// which are pairwise distinct for r = 0..15 -> conflict-free.
constexpr int LDSLD = 40;

// ---------------- helpers ---------------------------------------------------
__device__ __forceinline__ v8f vzero8() {
  v8f v;
#pragma unroll
  for (int i = 0; i < 8; ++i) v[i] = 0.0f;
  return v;
}

// A-matrix fragment, 16x32 f16 (ISA 7.12.2 "16-bit A-Matrix 16x32"):
// lanes 0-15 : row = lane,    VGPR0..3 = K 0..7,  VGPR4..7 = K 16..23
// lanes 16-31: row = lane-16, VGPR0..3 = K 8..15, VGPR4..7 = K 24..31
__device__ __forceinline__ v16h load_a16x32(const _Float16* __restrict__ S,
                                            int ld, int row0, int k0, int lane) {
  const int r   = row0 + (lane & 15);
  const int klo = k0 + ((lane & 16) ? 8 : 0);
  const v8h lo = *(const v8h*)(S + (size_t)r * ld + klo);
  const v8h hi = *(const v8h*)(S + (size_t)r * ld + klo + 16);
  v16h a;
#pragma unroll
  for (int i = 0; i < 8; ++i) { a[i] = lo[i]; a[8 + i] = hi[i]; }
  return a;
}

// B-matrix fragment, 32x16 f16: B(k,n) = W[col0+n][k] (W row-major [N][K]).
// lanes 0-15 hold K=k0..k0+7, k0+16..k0+23 for col=lane; lanes 16-31 hold K=k0+8..k0+15, k0+24..k0+31.
__device__ __forceinline__ v16h load_b32x16(const _Float16* __restrict__ W,
                                            int ld, int col0, int k0, int lane) {
  const int n  = col0 + (lane & 15);
  const int kb = k0 + ((lane & 16) ? 8 : 0);
  const v8h lo = *(const v8h*)(W + (size_t)n * ld + kb);
  const v8h hi = *(const v8h*)(W + (size_t)n * ld + kb + 16);
  v16h b;
#pragma unroll
  for (int i = 0; i < 8; ++i) { b[i] = lo[i]; b[8 + i] = hi[i]; }
  return b;
}

__device__ __forceinline__ v8f wmma16(v16h a, v16h b, v8f c) {
  // (neg_a, A, neg_b, B, c_mod, C, reuse_a, reuse_b)
  return __builtin_amdgcn_wmma_f32_16x16x32_f16(false, a, false, b,
                                                (short)0, c, false, false);
}

// ---------------- Tensor Data Mover: 2D f16 tile -> LDS ---------------------
// D# per ISA cdna5 §8.3/8.4. Tile of `rows` x 32 f16, source row stride
// `stride0` elements. Tensor dims == tile dims (all in-bounds, no OOB use).
// pad_enable: pad_interval code 3 (16 DWORDs = one 64B row),
// pad_amount code 3 (4 DWORDs = 16B) -> LDS row stride 80B (= LDSLD f16).
__device__ __forceinline__ void tdm_load_tile_f16(unsigned lds_off,
                                                  const _Float16* gptr,
                                                  unsigned rows,
                                                  unsigned long long stride0) {
  const unsigned long long ga = (unsigned long long)(uintptr_t)gptr;
  v4u g0;
  g0[0] = 1u;                                            // count=1, user mode
  g0[1] = lds_off;                                       // lds_addr (bytes)
  g0[2] = (unsigned)(ga & 0xFFFFFFFFu);                  // global_addr[31:0]
  g0[3] = (unsigned)((ga >> 32) & 0x01FFFFFFu)           // global_addr[56:32]
        | (2u << 30);                                    // type = 2 ("image")
  const unsigned td0 = 32u;                              // tile/tensor dim0
  const unsigned w0 = (1u << 16)                         // data_size = 2 bytes
                    | (1u << 20)                         // pad_enable
                    | (3u << 22)                         // pad_interval: 16 DW
                    | (3u << 25);                        // pad_amount: 4 DW
  v8i g1;
  g1[0] = (int)w0;
  g1[1] = (int)((td0 & 0xFFFFu) << 16);                  // tensor_dim0 lo16
  g1[2] = (int)((td0 >> 16) | ((rows & 0xFFFFu) << 16)); // td0 hi | td1 lo
  g1[3] = (int)((rows >> 16) | (td0 << 16));             // td1 hi | tile_dim0
  g1[4] = (int)(rows & 0xFFFFu);                         // tile_dim1 (dim2=0)
  g1[5] = (int)(unsigned)(stride0 & 0xFFFFFFFFu);        // dim0_stride lo32
  g1[6] = (int)(unsigned)((stride0 >> 32) & 0xFFFFu);    // dim0_stride hi16
  g1[7] = 0;
  const v4i z4 = {0, 0, 0, 0};
  const v8i z8 = {0, 0, 0, 0, 0, 0, 0, 0};
  __builtin_amdgcn_tensor_load_to_lds(g0, g1, z4, z4, z8, 0);
}

// ---------------- conversion / init kernels --------------------------------
__global__ void k_f32_to_f16(const float* __restrict__ in,
                             _Float16* __restrict__ out, size_t n) {
  size_t i = (size_t)blockIdx.x * blockDim.x + threadIdx.x;
  const size_t stride = (size_t)gridDim.x * blockDim.x;
  for (; i < n; i += stride) out[i] = (_Float16)in[i];
}

__global__ void k_init(const float* __restrict__ hx,
                       float* __restrict__ hf0, _Float16* __restrict__ hh0,
                       float* __restrict__ hf1, _Float16* __restrict__ hh1,
                       unsigned* __restrict__ cnt) {
  const int i = blockIdx.x * blockDim.x + threadIdx.x;
  if (i < BH_) {
    const float a = hx[i];        hf0[i] = a; hh0[i] = (_Float16)a;
    const float b = hx[BH_ + i];  hf1[i] = b; hh1[i] = (_Float16)b;
  }
  if (i < 2) cnt[i] = 0u;
}

// ---------------- big GEMM: GI[M,3H] = X[M,K] * W[3H,K]^T + bias ------------
// block = 256 threads (8 waves); block tile 128(M) x 256(N); wave tile 64x64.
// Per 32-wide K chunk, wave 0 issues two TDM loads (A: 128x32, B: 256x32)
// into double-buffered padded LDS, fenced by s_wait_tensorcnt + barrier, while
// all waves run 16 WMMAs on the previous chunk from LDS (ds_load_b128).
__global__ __launch_bounds__(256) void k_gemm_gi(const _Float16* __restrict__ X,
                                                 const _Float16* __restrict__ W,
                                                 const float* __restrict__ bias,
                                                 float* __restrict__ GI) {
  __shared__ _Float16 shA[2][128 * LDSLD];   // 2 x 10240 B
  __shared__ _Float16 shB[2][256 * LDSLD];   // 2 x 20480 B

  const int lane = threadIdx.x & 31;
  const int wave = threadIdx.x >> 5;
  const int mloc = (wave & 1) * 64;
  const int nloc = (wave >> 1) * 64;
  const int m0 = blockIdx.y * 128;
  const int n0 = blockIdx.x * 256;

  const _Float16* Ab = X + (size_t)m0 * KDIM;
  const _Float16* Bb = W + (size_t)n0 * KDIM;

  v8f acc[4][4];
#pragma unroll
  for (int i = 0; i < 4; ++i)
#pragma unroll
    for (int j = 0; j < 4; ++j) acc[i][j] = vzero8();

  // prologue: DMA chunk 0 into buffer 0
  if (wave == 0) {
    tdm_load_tile_f16((unsigned)(uintptr_t)&shA[0][0], Ab, 128u, KDIM);
    tdm_load_tile_f16((unsigned)(uintptr_t)&shB[0][0], Bb, 256u, KDIM);
    __builtin_amdgcn_s_wait_tensorcnt(0);
  }
  __syncthreads();

  constexpr int NC = KDIM / 32;
  for (int c = 0; c < NC; ++c) {
    const int buf = c & 1;
    if (wave == 0 && (c + 1) < NC) {
      const int k0n = (c + 1) * 32;
      tdm_load_tile_f16((unsigned)(uintptr_t)&shA[buf ^ 1][0], Ab + k0n, 128u, KDIM);
      tdm_load_tile_f16((unsigned)(uintptr_t)&shB[buf ^ 1][0], Bb + k0n, 256u, KDIM);
    }

    v16h a[4], b[4];
#pragma unroll
    for (int mt = 0; mt < 4; ++mt)
      a[mt] = load_a16x32(&shA[buf][0], LDSLD, mloc + mt * 16, 0, lane);
#pragma unroll
    for (int nt = 0; nt < 4; ++nt)
      b[nt] = load_b32x16(&shB[buf][0], LDSLD, nloc + nt * 16, 0, lane);
#pragma unroll
    for (int mt = 0; mt < 4; ++mt)
#pragma unroll
      for (int nt = 0; nt < 4; ++nt)
        acc[mt][nt] = wmma16(a[mt], b[nt], acc[mt][nt]);

    if (wave == 0 && (c + 1) < NC) __builtin_amdgcn_s_wait_tensorcnt(0);
    __syncthreads();
  }

  // C/D layout: element v -> row m0+mloc+16*mt+(lane>=16?8:0)+v,
  // col n0+nloc+16*nt+(lane&15)
  const int nl = lane & 15;
  const int mo = (lane & 16) ? 8 : 0;
#pragma unroll
  for (int nt = 0; nt < 4; ++nt) {
    const int n = n0 + nloc + nt * 16 + nl;
    const float bv = bias[n];
#pragma unroll
    for (int mt = 0; mt < 4; ++mt) {
#pragma unroll
      for (int v = 0; v < 8; ++v) {
        const int m = m0 + mloc + mt * 16 + mo + v;
        GI[(size_t)m * G3H + n] = acc[mt][nt][v] + bv;
      }
    }
  }
}

// ---------------- persistent recurrent kernel (one GRU layer) ---------------
// grid = 8 blocks x 256 threads (8 waves). Each wave owns 16 h-columns and
// computes gh_r/gh_z/gh_n for all 64 batch rows via 12 WMMAs per K-chunk.
// h is double buffered (f32 + f16 copies); a device-wide barrier (agent-scope
// atomics + s_sleep spin) separates timesteps. 8 blocks are always resident.
// gi (384 MB, streams from HBM) is prefetched at the top of each step so the
// gate epilogue hits cache.
__global__ __launch_bounds__(256) void k_gru_seq(
    const float* __restrict__ GI,      // [T,B,3H] precomputed input projection
    const _Float16* __restrict__ Whh,  // [3H,H] f16
    const float* __restrict__ bhh,     // [3H]
    float* __restrict__ hf,            // [2][B,H] f32 double buffer
    _Float16* __restrict__ hh,         // [2][B,H] f16 double buffer
    float* __restrict__ Y,             // [T,B,H] layer output
    float* __restrict__ Hn,            // [B,H] final hidden state
    unsigned* __restrict__ cnt, int T) {
  const int lane = threadIdx.x & 31;
  const int wave = threadIdx.x >> 5;
  const int n0 = (blockIdx.x * 8 + wave) * 16;   // 16 columns per wave
  const unsigned nb = gridDim.x;
  const int n = n0 + (lane & 15);
  const int mo = (lane & 16) ? 8 : 0;
  const float br_b = bhh[n];
  const float bz_b = bhh[H_ + n];
  const float bn_b = bhh[2 * H_ + n];

  for (int t = 0; t < T; ++t) {
    const int cur = t & 1;
    const _Float16* hr = hh + (size_t)cur * BH_;
    const float* git = GI + (size_t)t * B_ * G3H;

    // prefetch this step's gi cachelines (HBM) so epilogue reads hit cache;
    // lanes fan out over the 8 row-offsets within each 16-row tile.
    {
      const int pm = mo + (lane & 7);
#pragma unroll
      for (int mt = 0; mt < 4; ++mt) {
        const float* p = git + (size_t)(mt * 16 + pm) * G3H + n0;
        __builtin_prefetch(p, 0, 3);
        __builtin_prefetch(p + H_, 0, 3);
        __builtin_prefetch(p + 2 * H_, 0, 3);
      }
    }

    v8f ar[4], az[4], an_[4];
#pragma unroll
    for (int mt = 0; mt < 4; ++mt) { ar[mt] = vzero8(); az[mt] = vzero8(); an_[mt] = vzero8(); }

    for (int k0 = 0; k0 < H_; k0 += 32) {
      v16h a[4];
#pragma unroll
      for (int mt = 0; mt < 4; ++mt) a[mt] = load_a16x32(hr, H_, mt * 16, k0, lane);
      const v16h bfr = load_b32x16(Whh, H_, 0 * H_ + n0, k0, lane);
      const v16h bfz = load_b32x16(Whh, H_, 1 * H_ + n0, k0, lane);
      const v16h bfn = load_b32x16(Whh, H_, 2 * H_ + n0, k0, lane);
#pragma unroll
      for (int mt = 0; mt < 4; ++mt) {
        ar[mt]  = wmma16(a[mt], bfr, ar[mt]);
        az[mt]  = wmma16(a[mt], bfz, az[mt]);
        an_[mt] = wmma16(a[mt], bfn, an_[mt]);
      }
    }

    // fused gate epilogue
    float* yrow      = Y + (size_t)t * BH_;
    const float* hfc = hf + (size_t)cur * BH_;
    float* hfn       = hf + (size_t)(cur ^ 1) * BH_;
    _Float16* hhn    = hh + (size_t)(cur ^ 1) * BH_;
#pragma unroll
    for (int mt = 0; mt < 4; ++mt) {
#pragma unroll
      for (int v = 0; v < 8; ++v) {
        const int m = mt * 16 + mo + v;
        const size_t gbase = (size_t)m * G3H + n;
        const float gr  = git[gbase]           + ar[mt][v] + br_b;
        const float gz  = git[gbase + H_]      + az[mt][v] + bz_b;
        const float ghn =                        an_[mt][v] + bn_b;
        const float r = 1.0f / (1.0f + __expf(-gr));
        const float z = 1.0f / (1.0f + __expf(-gz));
        const float nn = tanhf(git[gbase + 2 * H_] + r * ghn);
        const float hold = hfc[m * H_ + n];
        const float hnew = (1.0f - z) * nn + z * hold;
        yrow[m * H_ + n] = hnew;
        hfn[m * H_ + n]  = hnew;
        hhn[m * H_ + n]  = (_Float16)hnew;
        if (t == T - 1) Hn[m * H_ + n] = hnew;
      }
    }

    // device-wide barrier before the next timestep reads hh[cur^1]
    if (t + 1 < T) {
      __syncthreads();
      if (threadIdx.x == 0) {
        __hip_atomic_fetch_add(cnt, 1u, __ATOMIC_RELEASE, __HIP_MEMORY_SCOPE_AGENT);
        const unsigned target = nb * (unsigned)(t + 1);
        while (__hip_atomic_load(cnt, __ATOMIC_ACQUIRE, __HIP_MEMORY_SCOPE_AGENT) < target) {
          __builtin_amdgcn_s_sleep(2);
        }
      }
      __syncthreads();
    }
  }
}

// ---------------- host launcher ---------------------------------------------
extern "C" void kernel_launch(void* const* d_in, const int* in_sizes, int n_in,
                              void* d_out, int out_size, void* d_ws, size_t ws_size,
                              hipStream_t stream) {
  (void)in_sizes; (void)n_in; (void)out_size; (void)ws_size;
  const float* x      = (const float*)d_in[0];
  const float* hx     = (const float*)d_in[1];
  const float* w_ih_0 = (const float*)d_in[2];
  const float* w_hh_0 = (const float*)d_in[3];
  const float* b_ih_0 = (const float*)d_in[4];
  const float* b_hh_0 = (const float*)d_in[5];
  const float* w_ih_1 = (const float*)d_in[6];
  const float* w_hh_1 = (const float*)d_in[7];
  const float* b_ih_1 = (const float*)d_in[8];
  const float* b_hh_1 = (const float*)d_in[9];

  float* out = (float*)d_out;
  float* y1  = out;                                   // [T,B,H]
  float* hn  = out + (size_t)T_ * BH_;                // [2,B,H]

  const size_t TB = (size_t)T_ * B_;                  // 32768

  // workspace carve-up
  char* ws = (char*)d_ws;
  size_t off = 0;
  auto carve = [&](size_t bytes) -> char* {
    char* p = ws + off;
    off = (off + bytes + 255) & ~(size_t)255;
    return p;
  };
  float*    GI    = (float*)    carve(TB * G3H * sizeof(float));       // 384 MiB
  _Float16* XH    = (_Float16*) carve(TB * KDIM * sizeof(_Float16));   //  64 MiB
  float*    Y0    = (float*)    carve(TB * H_ * sizeof(float));        // 128 MiB
  _Float16* Wih0h = (_Float16*) carve((size_t)G3H * KDIM * 2);
  _Float16* Whh0h = (_Float16*) carve((size_t)G3H * KDIM * 2);
  _Float16* Wih1h = (_Float16*) carve((size_t)G3H * KDIM * 2);
  _Float16* Whh1h = (_Float16*) carve((size_t)G3H * KDIM * 2);
  float*    hf0   = (float*)    carve(2 * (size_t)BH_ * sizeof(float));
  _Float16* hh0   = (_Float16*) carve(2 * (size_t)BH_ * sizeof(_Float16));
  float*    hf1   = (float*)    carve(2 * (size_t)BH_ * sizeof(float));
  _Float16* hh1   = (_Float16*) carve(2 * (size_t)BH_ * sizeof(_Float16));
  unsigned* cnts  = (unsigned*) carve(256);

  // 1) init hidden state + barrier counters
  k_init<<<(BH_ + 255) / 256, 256, 0, stream>>>(hx, hf0, hh0, hf1, hh1, cnts);

  // 2) f32 -> f16 conversions
  k_f32_to_f16<<<4096, 256, 0, stream>>>(x,      XH,    TB * (size_t)I_);
  k_f32_to_f16<<<1024, 256, 0, stream>>>(w_ih_0, Wih0h, (size_t)G3H * KDIM);
  k_f32_to_f16<<<1024, 256, 0, stream>>>(w_hh_0, Whh0h, (size_t)G3H * KDIM);
  k_f32_to_f16<<<1024, 256, 0, stream>>>(w_ih_1, Wih1h, (size_t)G3H * KDIM);
  k_f32_to_f16<<<1024, 256, 0, stream>>>(w_hh_1, Whh1h, (size_t)G3H * KDIM);

  const dim3 gemmGrid(G3H / 256, (unsigned)(TB / 128));  // (12, 256)

  // 3) layer 0: input projection + sequential recurrence
  k_gemm_gi<<<gemmGrid, 256, 0, stream>>>(XH, Wih0h, b_ih_0, GI);
  k_gru_seq<<<H_ / 128, 256, 0, stream>>>(GI, Whh0h, b_hh_0, hf0, hh0,
                                          Y0, hn /* h_n[0] */, cnts + 0, T_);

  // 4) layer 1: y0 -> f16, input projection, sequential recurrence
  k_f32_to_f16<<<4096, 256, 0, stream>>>(Y0, XH, TB * (size_t)H_);
  k_gemm_gi<<<gemmGrid, 256, 0, stream>>>(XH, Wih1h, b_ih_1, GI);
  k_gru_seq<<<H_ / 128, 256, 0, stream>>>(GI, Whh1h, b_hh_1, hf1, hh1,
                                          y1, hn + BH_ /* h_n[1] */, cnts + 1, T_);
}
